// Model_50208167690291
// MI455X (gfx1250) — matrix operand order repaired and checked by reading, not compiled, not run on hardware
//
#include <hip/hip_runtime.h>
#include <math.h>

// ---------------------------------------------------------------------------
// Informer forward for MI455X (gfx1250, wave32).
// Dense layers: v_wmma_f32_16x16x32_f16 with A staged via async-to-LDS.
// ---------------------------------------------------------------------------

typedef __attribute__((ext_vector_type(16))) _Float16 v16h;
typedef __attribute__((ext_vector_type(8)))  _Float16 v8h;
typedef __attribute__((ext_vector_type(8)))  float    v8f;
typedef __attribute__((address_space(3))) _Float16* as3h;

static constexpr int BS   = 8;    // batch
static constexpr int DMO  = 512;  // d_model
static constexpr int NH   = 8;    // heads
static constexpr int DH   = 64;   // head dim
static constexpr int UQ   = 28;   // top-k queries (4*ceil(ln L)), same for all L here
static constexpr int UK   = 28;   // sampled keys

__device__ __forceinline__ v16h cat8(v8h lo, v8h hi) {
  return __builtin_shufflevector(lo, hi, 0,1,2,3,4,5,6,7,8,9,10,11,12,13,14,15);
}

// ---------------- WMMA GEMM:  Y[M,N] = A[M,K](f16) * Wt[K,N](f16) + bias ----
// Wt pre-transposed to K-major [K,N]; B-frag elements are single f16 loads at stride N.
// Block: 8 waves x 64 N-cols = N=512 wide; MT (1 or 2, compile-time) 16-row
// M tiles. A tile (rows x K halfs, row stride K+8 for bank stagger) is staged
// into LDS once per block with GLOBAL_LOAD_ASYNC_TO_LDS_B128 (ASYNCcnt), then
// A-frags come from LDS (ds_load_b128), reused by all 8 waves and both tiles.
// A-frag (16-bit A 16x32): lane m=lane&15, h=lane>>4; j<8 -> K=8h+j,
// j>=8 -> K=16+8h+(j-8). B-frag: N=lane&15, slot j uses the same K as A. C/D: N=lane&15, M=r+8*(lane>>4).
template<int MT>
__global__ void __launch_bounds__(256) gemm_wmma(
    const _Float16* __restrict__ A, const _Float16* __restrict__ Wt,
    const float* __restrict__ bias, float* __restrict__ Y,
    int M, int N, int K, int act)
{
  extern __shared__ _Float16 sA[];   // (16*MT) x (K+8)
  const int tid    = threadIdx.x;
  const int rows   = MT << 4;
  const int stride = K + 8;
  const int m0     = blockIdx.y * rows;

  // ---- async-stage A tile: rows x K halfs, 16B chunks, all 256 threads ----
  const int cpr    = K >> 3;          // 16B chunks per row
  const int chunks = rows * cpr;      // multiple of 256 for all shapes here
  for (int c = tid; c < chunks; c += 256) {
    const int row = c / cpr;
    const int col = (c - row * cpr) << 3;
    const _Float16* g = A + (size_t)(m0 + row) * K + col;
    unsigned loff = (unsigned)(size_t)(as3h)(sA + (size_t)row * stride + col);
    asm volatile("global_load_async_to_lds_b128 %0, %1, off"
                 :: "v"(loff), "v"((unsigned long long)(size_t)g) : "memory");
  }
  asm volatile("s_wait_asynccnt 0x0" ::: "memory");
  __syncthreads();

  const int wave = tid >> 5;
  const int lane = tid & 31;
  const int mr = lane & 15;
  const int hs = lane >> 4;
  const int n0 = (blockIdx.x * 8 + wave) << 6;
  const _Float16* Bc = Wt + (size_t)(8 * hs) * N + n0 + mr;
  const _Float16* s0b = sA + (size_t)mr * stride + 8 * hs;

  v8f acc[MT][4] = {};
  for (int k0 = 0; k0 < K; k0 += 32) {
    const _Float16* s0 = s0b + k0;
    v16h a[MT];
    a[0] = cat8(*(const v8h*)(s0), *(const v8h*)(s0 + 16));
    if (MT == 2) {
      const _Float16* s1 = s0 + (size_t)16 * stride;
      a[1] = cat8(*(const v8h*)(s1), *(const v8h*)(s1 + 16));
    }
    const _Float16* Bp = Bc + (size_t)k0 * N;
    v16h b0, b1, b2, b3;
    #pragma unroll
    for (int s = 0; s < 16; ++s) {
      const size_t ko = (size_t)((s < 8) ? s : s + 8) * N;
      b0[s] = Bp[ko]; b1[s] = Bp[ko + 16]; b2[s] = Bp[ko + 32]; b3[s] = Bp[ko + 48];
    }
    if (k0 + 32 < K)   // gfx1250 global_prefetch_b8
      __builtin_prefetch((const void*)(Bp + (size_t)32 * N), 0, 3);
    #pragma unroll
    for (int tl = 0; tl < MT; ++tl) {
      acc[tl][0] = __builtin_amdgcn_wmma_f32_16x16x32_f16(false, a[tl], false, b0, (short)0, acc[tl][0], false, false);
      acc[tl][1] = __builtin_amdgcn_wmma_f32_16x16x32_f16(false, a[tl], false, b1, (short)0, acc[tl][1], false, false);
      acc[tl][2] = __builtin_amdgcn_wmma_f32_16x16x32_f16(false, a[tl], false, b2, (short)0, acc[tl][2], false, false);
      acc[tl][3] = __builtin_amdgcn_wmma_f32_16x16x32_f16(false, a[tl], false, b3, (short)0, acc[tl][3], false, false);
    }
  }
  #pragma unroll
  for (int tl = 0; tl < MT; ++tl) {
    #pragma unroll
    for (int t = 0; t < 4; ++t) {
      const int n = n0 + t*16 + mr;
      const float bv = bias ? bias[n] : 0.0f;
      #pragma unroll
      for (int r = 0; r < 8; ++r) {
        const int m = m0 + tl*16 + r + 8*hs;
        float v = acc[tl][t][r] + bv;
        if (act == 1) v = 0.5f * v * (1.0f + erff(v * 0.70710678118654752f)); // exact GELU
        Y[(size_t)m * N + n] = v;
      }
    }
  }
}

// ---------------- elementwise / conversion kernels --------------------------
__global__ void f16cast(const float* __restrict__ x, _Float16* __restrict__ y, size_t n) {
  size_t i = (size_t)blockIdx.x * blockDim.x + threadIdx.x;
  if (i < n) y[i] = (_Float16)x[i];
}

// w[N,K] f32 -> wt[K,N] f16
__global__ void wtransp(const float* __restrict__ w, _Float16* __restrict__ wt, int N, int K) {
  size_t i = (size_t)blockIdx.x * blockDim.x + threadIdx.x;
  if (i >= (size_t)N * K) return;
  int k = (int)(i % K), n = (int)(i / K);
  wt[(size_t)k * N + n] = (_Float16)w[i];
}

// conv weight (O=512, C=512, 3) -> wt[(kidx*512+c), o] f16 (K-major for im2col GEMM)
__global__ void convw(const float* __restrict__ w, _Float16* __restrict__ wt) {
  size_t j = (size_t)blockIdx.x * blockDim.x + threadIdx.x;
  if (j >= (size_t)3 * DMO * DMO) return;
  int o  = (int)(j % DMO);
  int c  = (int)((j / DMO) % DMO);
  int kk = (int)(j / ((size_t)DMO * DMO));
  wt[j] = (_Float16)w[(size_t)o * (DMO*3) + c*3 + kk];
}

// im2col for circular conv3: A16[tok, kidx*512 + c] = x[(l+kidx-1) mod L, c]
__global__ void im2col3(const float* __restrict__ X, _Float16* __restrict__ A16, int L) {
  size_t i = (size_t)blockIdx.x * blockDim.x + threadIdx.x;
  size_t total = (size_t)BS * L * 3 * DMO;
  if (i >= total) return;
  int c    = (int)(i % DMO);
  int kidx = (int)((i / DMO) % 3);
  size_t tok = i / ((size_t)3 * DMO);
  int l = (int)(tok % L), b = (int)(tok / L);
  int lm = l + kidx - 1; lm = (lm + L) % L;
  A16[i] = (_Float16)X[((size_t)(b*L + lm)) * DMO + c];
}

// embedding: circular conv3 (cin=7) + temporal emb + sinusoidal pos emb
__global__ void embed_kernel(const float* __restrict__ x, const int* __restrict__ mark,
                             const float* __restrict__ tokw, const float* __restrict__ month,
                             const float* __restrict__ day,  const float* __restrict__ wday,
                             const float* __restrict__ hour, float* __restrict__ out, int L)
{
  size_t i = (size_t)blockIdx.x * blockDim.x + threadIdx.x;
  if (i >= (size_t)BS * L * DMO) return;
  int o = (int)(i % DMO);
  int l = (int)((i / DMO) % L);
  int b = (int)(i / ((size_t)DMO * L));
  float s = 0.f;
  #pragma unroll
  for (int k = 0; k < 3; ++k) {
    int lm = l + k - 1; lm = (lm + L) % L;
    const float* xr = x + ((size_t)(b*L + lm)) * 7;
    #pragma unroll
    for (int c = 0; c < 7; ++c) s += xr[c] * tokw[(size_t)o*21 + c*3 + k];
  }
  const int* mk = mark + ((size_t)(b*L + l)) * 4;
  s += month[(size_t)mk[0]*DMO + o] + day[(size_t)mk[1]*DMO + o]
     + wday [(size_t)mk[2]*DMO + o] + hour[(size_t)mk[3]*DMO + o];
  float div = expf(-9.210340371976184f * (float)(o & ~1) / (float)DMO);
  float ang = (float)l * div;
  s += (o & 1) ? cosf(ang) : sinf(ang);
  out[i] = s;
}

// ---------------- reductions ------------------------------------------------
__device__ __forceinline__ float blk_sum(float v, float* red) {
  int t = threadIdx.x; red[t] = v; __syncthreads();
  #pragma unroll
  for (int s = 128; s > 0; s >>= 1) { if (t < s) red[t] += red[t+s]; __syncthreads(); }
  float r = red[0]; __syncthreads(); return r;
}
__device__ __forceinline__ float blk_max(float v, float* red) {
  int t = threadIdx.x; red[t] = v; __syncthreads();
  #pragma unroll
  for (int s = 128; s > 0; s >>= 1) { if (t < s) red[t] = fmaxf(red[t], red[t+s]); __syncthreads(); }
  float r = red[0]; __syncthreads(); return r;
}

// LayerNorm(x [+ r]) over D=512; one block per token; in-place safe
__global__ void __launch_bounds__(256) ln_res(const float* __restrict__ X,
                                              const float* __restrict__ R,
                                              const float* __restrict__ G,
                                              const float* __restrict__ Bt,
                                              float* __restrict__ Y)
{
  __shared__ float red[256];
  const size_t t = blockIdx.x;
  const int i0 = threadIdx.x, i1 = threadIdx.x + 256;
  const float* xr = X + t * DMO;
  float v0 = xr[i0], v1 = xr[i1];
  if (R) { const float* rr = R + t * DMO; v0 += rr[i0]; v1 += rr[i1]; }
  float mean = blk_sum(v0 + v1, red) * (1.0f / DMO);
  float d0 = v0 - mean, d1 = v1 - mean;
  float var = blk_sum(d0*d0 + d1*d1, red) * (1.0f / DMO);
  float rstd = rsqrtf(var + 1e-5f);
  float* yr = Y + t * DMO;
  yr[i0] = d0 * rstd * G[i0] + Bt[i0];
  yr[i1] = d1 * rstd * G[i1] + Bt[i1];
}

// ---------------- ProbSparse attention pieces -------------------------------
__device__ __forceinline__ int sample_idx(unsigned seed, int l, int s, int Lk) {
  unsigned a = seed * 0x9E3779B9u + (unsigned)(l * UK + s);
  a ^= a >> 16; a *= 0x7feb352du; a ^= a >> 15; a *= 0x846ca68bu; a ^= a >> 16;
  return (int)(a & (unsigned)(Lk - 1));  // Lk is a power of two (512/1024)
}

// M[l] = max_s q.k_s - sum_s/Lk over sampled keys  (sparsity measure)
__global__ void attn_measure(const float* __restrict__ Q, const float* __restrict__ Kf,
                             float* __restrict__ Mout, int Lq, int Lk, unsigned seed)
{
  int bh = blockIdx.y, b = bh / NH, hh = bh % NH;
  int l = blockIdx.x * blockDim.x + threadIdx.x;
  if (l >= Lq) return;
  const float4* q4 = (const float4*)(Q + ((size_t)(b*Lq + l)) * DMO + hh*DH);
  float4 qv[16];
  #pragma unroll
  for (int i = 0; i < 16; ++i) qv[i] = q4[i];
  float mx = -1e30f, sm = 0.f;
  for (int s = 0; s < UK; ++s) {
    int j = sample_idx(seed, l, s, Lk);
    const float4* k4 = (const float4*)(Kf + ((size_t)(b*Lk + j)) * DMO + hh*DH);
    float d = 0.f;
    #pragma unroll
    for (int i = 0; i < 16; ++i) {
      float4 kv = k4[i];
      d += qv[i].x*kv.x + qv[i].y*kv.y + qv[i].z*kv.z + qv[i].w*kv.w;
    }
    mx = fmaxf(mx, d); sm += d;
  }
  Mout[(size_t)bh * Lq + l] = mx - sm / (float)Lk;
}

// top-28 indices per (b,h) via iterative argmax in LDS (ties -> lower index)
__global__ void __launch_bounds__(256) topk28(const float* __restrict__ Mv,
                                              int* __restrict__ top, int Lq)
{
  __shared__ float mv[1024];
  __shared__ float smax[256];
  __shared__ int   sidx[256];
  int bh = blockIdx.x, tid = threadIdx.x;
  for (int i = tid; i < Lq; i += 256) mv[i] = Mv[(size_t)bh * Lq + i];
  __syncthreads();
  for (int u = 0; u < UQ; ++u) {
    float bm = -1e30f; int bi = 0;
    for (int i = tid; i < Lq; i += 256) { float v = mv[i]; if (v > bm) { bm = v; bi = i; } }
    smax[tid] = bm; sidx[tid] = bi; __syncthreads();
    #pragma unroll
    for (int s = 128; s > 0; s >>= 1) {
      if (tid < s) {
        if (smax[tid+s] > smax[tid] || (smax[tid+s] == smax[tid] && sidx[tid+s] < sidx[tid])) {
          smax[tid] = smax[tid+s]; sidx[tid] = sidx[tid+s];
        }
      }
      __syncthreads();
    }
    if (tid == 0) { top[(size_t)bh * UQ + u] = sidx[0]; mv[sidx[0]] = -1e30f; }
    __syncthreads();
  }
}

// ctx = broadcast mean of V over keys (non-masked branch)
__global__ void ctx_mean(const float* __restrict__ V, float* __restrict__ C, int Lq, int Lk) {
  int bh = blockIdx.x, b = bh / NH, hh = bh % NH, d = threadIdx.x;
  float s = 0.f;
  for (int l = 0; l < Lk; ++l) s += V[((size_t)(b*Lk + l)) * DMO + hh*DH + d];
  s /= (float)Lk;
  for (int l = 0; l < Lq; ++l) C[((size_t)(b*Lq + l)) * DMO + hh*DH + d] = s;
}

// ctx = cumulative sum of V (masked branch, Lq == Lk)
__global__ void ctx_cumsum(const float* __restrict__ V, float* __restrict__ C, int L) {
  int bh = blockIdx.x, b = bh / NH, hh = bh % NH, d = threadIdx.x;
  float s = 0.f;
  for (int l = 0; l < L; ++l) {
    s += V[((size_t)(b*L + l)) * DMO + hh*DH + d];
    C[((size_t)(b*L + l)) * DMO + hh*DH + d] = s;
  }
}

// full attention for the selected queries, scatter into ctx
__global__ void __launch_bounds__(256) attn_selected(
    const float* __restrict__ Q, const float* __restrict__ Kf, const float* __restrict__ V,
    const int* __restrict__ top, float* __restrict__ Ctx, int Lq, int Lk, int masked)
{
  __shared__ float sc[1024];
  __shared__ float qs[DH];
  __shared__ float red[256];
  __shared__ float outv[DH];
  const int u = blockIdx.x, bh = blockIdx.y, b = bh / NH, hh = bh % NH;
  const int tid = threadIdx.x;
  const int qi = top[(size_t)bh * UQ + u];
  if (tid < DH) qs[tid] = Q[((size_t)(b*Lq + qi)) * DMO + hh*DH + tid];
  __syncthreads();
  const int lim = masked ? (qi + 1) : Lk;   // mask: key index > query position
  float mx = -1e30f;
  for (int j = tid; j < Lk; j += 256) {
    float s = -1e30f;
    if (j < lim) {
      s = 0.f;
      const float* kp = Kf + ((size_t)(b*Lk + j)) * DMO + hh*DH;
      #pragma unroll
      for (int i = 0; i < DH; ++i) s += qs[i] * kp[i];
      s *= 0.125f;  // 1/sqrt(64)
    }
    sc[j] = s;
    mx = fmaxf(mx, s);
  }
  float gmx = blk_max(mx, red);
  float ls = 0.f;
  for (int j = tid; j < Lk; j += 256) { float e = __expf(sc[j] - gmx); sc[j] = e; ls += e; }
  float tot = blk_sum(ls, red);
  __syncthreads();
  if (tid < DH) {
    float acc = 0.f;
    for (int j = 0; j < lim; ++j) acc += sc[j] * V[((size_t)(b*Lk + j)) * DMO + hh*DH + tid];
    outv[tid] = acc / tot;
  }
  __syncthreads();
  if (tid < DH) Ctx[((size_t)(b*Lq + qi)) * DMO + hh*DH + tid] = outv[tid];
}

// ---------------- conv distilling layer pieces ------------------------------
__global__ void bn_elu(float* __restrict__ Y, const float* __restrict__ G,
                       const float* __restrict__ Bb, size_t n)
{
  size_t i = (size_t)blockIdx.x * blockDim.x + threadIdx.x;
  if (i >= n) return;
  int c = (int)(i % DMO);
  float v = Y[i] * 0.9999950000374996f * G[c] + Bb[c];   // y/sqrt(1+1e-5)*g + b
  Y[i] = v > 0.f ? v : (expf(v) - 1.f);                  // ELU
}

__global__ void maxpool3(const float* __restrict__ Yin, float* __restrict__ Out, int Lin) {
  int Lout = Lin >> 1;
  size_t i = (size_t)blockIdx.x * blockDim.x + threadIdx.x;
  if (i >= (size_t)BS * Lout * DMO) return;
  int c = (int)(i % DMO);
  int l = (int)((i / DMO) % Lout);
  int b = (int)(i / ((size_t)DMO * Lout));
  float m = -1e30f;
  #pragma unroll
  for (int dl = -1; dl <= 1; ++dl) {
    int ll = 2*l + dl;
    if (ll >= 0 && ll < Lin) m = fmaxf(m, Yin[((size_t)(b*Lin + ll)) * DMO + c]);
  }
  Out[((size_t)(b*Lout + l)) * DMO + c] = m;
}

// final projection 512 -> 7, taking the last 512 positions of the 1024 decoder tokens
__global__ void proj_out_k(const float* __restrict__ X, const float* __restrict__ W,
                           const float* __restrict__ Bsv, float* __restrict__ Out)
{
  int i = blockIdx.x * blockDim.x + threadIdx.x;
  if (i >= BS * 512 * 7) return;
  int c = i % 7, l = (i / 7) % 512, b = i / (7 * 512);
  const float* xr = X + ((size_t)(b*1024 + 512 + l)) * DMO;
  const float* wr = W + (size_t)c * DMO;
  float s = Bsv[c];
  for (int k = 0; k < DMO; ++k) s += xr[k] * wr[k];
  Out[i] = s;
}

// ---------------------------------------------------------------------------
// Host orchestration.
// Param pytree (dicts sorted by key) flattens to d_in[4..81]:
//  dec_emb: day=4 hour=5 month=6 tok_w=7 wday=8
//  dec_layers[0]: c1b=9 c1w=10 c2b=11 c2w=12
//    cattn: bk=13 bo=14 bq=15 bv=16 wk=17 wo=18 wq=19 wv=20
//    n1b=21 n1g=22 n2b=23 n2g=24 n3b=25 n3g=26
//    sattn: bk=27 bo=28 bq=29 bv=30 wk=31 wo=32 wq=33 wv=34
//  dec_nb=35 dec_ng=36
//  enc_convs[0]: b=37 bb=38 g=39 w=40
//  enc_emb: day=41 hour=42 month=43 tok_w=44 wday=45
//  enc_layers[0]: attn bk=46 bo=47 bq=48 bv=49 wk=50 wo=51 wq=52 wv=53
//                 c1b=54 c1w=55 c2b=56 c2w=57 n1b=58 n1g=59 n2b=60 n2g=61
//  enc_layers[1]: attn 62..69, c1b=70 c1w=71 c2b=72 c2w=73 n1b=74 n1g=75 n2b=76 n2g=77
//  enc_nb=78 enc_ng=79 proj_b=80 proj_w=81
// ---------------------------------------------------------------------------
extern "C" void kernel_launch(void* const* d_in, const int* in_sizes, int n_in,
                              void* d_out, int out_size, void* d_ws, size_t ws_size,
                              hipStream_t stream)
{
  (void)in_sizes; (void)n_in; (void)out_size; (void)ws_size;
  const float* x_enc    = (const float*)d_in[0];
  const int*   mark_enc = (const int*)  d_in[1];
  const float* x_dec    = (const float*)d_in[2];
  const int*   mark_dec = (const int*)  d_in[3];
  auto P = [&](int i) { return (const float*)d_in[i]; };

  // ---- workspace layout (recomputed identically every call) ----
  char* ws = (char*)d_ws;
  size_t off = 0;
  auto alloc = [&](size_t b) { size_t r = off; off += (b + 255) & ~(size_t)255; return r; };
  const size_t ACT32 = (size_t)8192 * DMO * 4;
  float*    X    = (float*)(ws + alloc(ACT32));            // encoder activations
  float*    DXb  = (float*)(ws + alloc(ACT32));            // decoder activations
  float*    QB   = (float*)(ws + alloc(ACT32));
  float*    KB   = (float*)(ws + alloc(ACT32));
  float*    VB   = (float*)(ws + alloc(ACT32));
  float*    CB   = (float*)(ws + alloc(ACT32));
  _Float16* H16  = (_Float16*)(ws + alloc((size_t)8192 * DMO * 2));
  _Float16* H16E = (_Float16*)(ws + alloc((size_t)4096 * DMO * 2));
  float*    MB   = (float*)(ws + alloc((size_t)BS * NH * 1024 * 4));
  int*      TOPB = (int*)  (ws + alloc((size_t)BS * NH * UQ * 4));
  _Float16* WT[23];
  for (int i = 0; i < 22; ++i) WT[i] = (_Float16*)(ws + alloc((size_t)DMO * DMO * 2));
  WT[22] = (_Float16*)(ws + alloc((size_t)3 * DMO * DMO * 2));   // conv im2col weight
  _Float16* A16 = (_Float16*)QB;   // im2col aliases QB+KB (conv runs when both are free)

  // ---- pre-transpose all GEMM weights to K-major f16 ----
  // [0..5] enc0 q,k,v,o,c1,c2  [6..11] enc1  [12..15] dec-self  [16..19] dec-cross  [20,21] dec ffn
  const int widx[22] = {52,50,53,51,55,57, 68,66,69,67,71,73, 33,31,34,32, 19,17,20,18, 10,12};
  for (int i = 0; i < 22; ++i)
    wtransp<<<dim3(1024), 256, 0, stream>>>(P(widx[i]), WT[i], DMO, DMO);
  convw<<<dim3(3072), 256, 0, stream>>>(P(40), WT[22]);

  auto cast16 = [&](const float* src, _Float16* dst, size_t n) {
    f16cast<<<dim3((unsigned)((n + 255) / 256)), 256, 0, stream>>>(src, dst, n);
  };
  auto gemm = [&](const _Float16* A, const _Float16* Wt, const float* bias,
                  float* Y, int M, int K, int act) {
    if (K > 1024) {  // conv GEMM: 16 rows to keep dynamic LDS < 64KB
      const size_t sh = (size_t)16 * (K + 8) * sizeof(_Float16);
      gemm_wmma<1><<<dim3(1, M / 16), 256, sh, stream>>>(A, Wt, bias, Y, M, DMO, K, act);
    } else {         // projection/FFN GEMMs: 32 rows per block
      const size_t sh = (size_t)32 * (K + 8) * sizeof(_Float16);
      gemm_wmma<2><<<dim3(1, M / 32), 256, sh, stream>>>(A, Wt, bias, Y, M, DMO, K, act);
    }
  };

  // ProbSparse attention block. Output (after O-projection) lands in QB.
  auto attention = [&](float* Xq, int Mq, int Lq, const float* Xkv, int Mkv, int Lk,
                       int wb, const float* bq, const float* bk, const float* bv,
                       const float* bo, int masked, unsigned seed) {
    cast16(Xq, H16, (size_t)Mq * DMO);
    gemm(H16, WT[wb+0], bq, QB, Mq, DMO, 0);
    const _Float16* HK = H16;
    if (Xkv != (const float*)Xq) { cast16(Xkv, H16E, (size_t)Mkv * DMO); HK = H16E; }
    gemm(HK, WT[wb+1], bk, KB, Mkv, DMO, 0);
    gemm(HK, WT[wb+2], bv, VB, Mkv, DMO, 0);
    attn_measure<<<dim3((Lq + 255) / 256, BS * NH), 256, 0, stream>>>(QB, KB, MB, Lq, Lk, seed);
    topk28<<<dim3(BS * NH), 256, 0, stream>>>(MB, TOPB, Lq);
    if (masked) ctx_cumsum<<<dim3(BS * NH), DH, 0, stream>>>(VB, CB, Lq);
    else        ctx_mean  <<<dim3(BS * NH), DH, 0, stream>>>(VB, CB, Lq, Lk);
    attn_selected<<<dim3(UQ, BS * NH), 256, 0, stream>>>(QB, KB, VB, TOPB, CB, Lq, Lk, masked);
    cast16(CB, H16, (size_t)Mq * DMO);
    gemm(H16, WT[wb+3], bo, QB, Mq, DMO, 0);
  };

  auto ffn = [&](float* Xio, int Mtok, int wc1, const float* c1b, int wc2,
                 const float* c2b, const float* ng, const float* nb) {
    cast16(Xio, H16, (size_t)Mtok * DMO);
    gemm(H16, WT[wc1], c1b, KB, Mtok, DMO, 1 /*gelu*/);
    cast16(KB, H16, (size_t)Mtok * DMO);
    gemm(H16, WT[wc2], c2b, VB, Mtok, DMO, 0);
    ln_res<<<dim3(Mtok), 256, 0, stream>>>(Xio, VB, ng, nb, Xio);
  };

  // ---- embeddings (x_dec_in == x_dec, x_mark == x_mark_dec since Ld == len) ----
  embed_kernel<<<dim3(16384), 256, 0, stream>>>(x_enc, mark_enc, P(44), P(43), P(41), P(45), P(42), X,   1024);
  embed_kernel<<<dim3(16384), 256, 0, stream>>>(x_dec, mark_dec, P(7),  P(6),  P(4),  P(8),  P(5),  DXb, 1024);

  // ---- encoder layer 0 (L=1024, 8192 tokens) ----
  attention(X, 8192, 1024, X, 8192, 1024, 0, P(48), P(46), P(49), P(47), 0, 0u);
  ln_res<<<dim3(8192), 256, 0, stream>>>(X, QB, P(59), P(58), X);
  ffn(X, 8192, 4, P(54), 5, P(56), P(61), P(60));

  // ---- conv distilling layer: circ conv3 as im2col GEMM, BN-ish, ELU, maxpool ----
  im2col3<<<dim3(49152), 256, 0, stream>>>(X, A16, 1024);
  gemm(A16, WT[22], P(37), CB, 8192, 3 * DMO, 0);
  bn_elu<<<dim3(16384), 256, 0, stream>>>(CB, P(39), P(38), (size_t)8192 * DMO);
  maxpool3<<<dim3(8192), 256, 0, stream>>>(CB, X, 1024);   // -> 4096 tokens in X

  // ---- encoder layer 1 (L=512, 4096 tokens) ----
  attention(X, 4096, 512, X, 4096, 512, 6, P(64), P(62), P(65), P(63), 0, 1u);
  ln_res<<<dim3(4096), 256, 0, stream>>>(X, QB, P(75), P(74), X);
  ffn(X, 4096, 10, P(70), 11, P(72), P(77), P(76));
  ln_res<<<dim3(4096), 256, 0, stream>>>(X, nullptr, P(79), P(78), X);  // encoder norm

  // ---- decoder layer (L=1024, 8192 tokens) ----
  attention(DXb, 8192, 1024, DXb, 8192, 1024, 12, P(29), P(27), P(30), P(28), 1, 100u); // masked self
  ln_res<<<dim3(8192), 256, 0, stream>>>(DXb, QB, P(22), P(21), DXb);
  attention(DXb, 8192, 1024, X, 4096, 512, 16, P(15), P(13), P(16), P(14), 0, 101u);    // cross
  ln_res<<<dim3(8192), 256, 0, stream>>>(DXb, QB, P(24), P(23), DXb);
  ffn(DXb, 8192, 20, P(9), 21, P(11), P(26), P(25));
  ln_res<<<dim3(8192), 256, 0, stream>>>(DXb, nullptr, P(36), P(35), DXb);              // decoder norm

  // ---- projection to 7 channels, last 512 positions ----
  proj_out_k<<<dim3(112), 256, 0, stream>>>(DXb, P(81), P(80), (float*)d_out);
}
